// PIGNN_Hybrid_29669634081215
// MI455X (gfx1250) — compile-verified, matrix operand order checked by reading, not run on hardware
//
#include <hip/hip_runtime.h>
#include <cstdint>
#include <cstddef>

// ---------------------------------------------------------------------------
// PIGNN hybrid pipeline for MI455X (gfx1250, wave32, WMMA).
// All dense math: v_wmma_f32_16x16x32_bf16 (bf16 in, f32 accum).
// Activations and weights are kept in bf16 between GEMMs so LDS staging is a
// pure 32-bit copy; f32 is kept only where numerics need it (LN residual,
// atomic scatter-add) and for raw inputs.
// ---------------------------------------------------------------------------

typedef float v8f  __attribute__((ext_vector_type(8)));
typedef __bf16 v16bf __attribute__((ext_vector_type(16)));
typedef unsigned int u32x4 __attribute__((ext_vector_type(4)));

union FragBF { v16bf v; u32x4 q[2]; };

__device__ __forceinline__ unsigned short f2bf(float f) {
    unsigned u = __float_as_uint(f);
    u += 0x7FFFu + ((u >> 16) & 1u);          // round-to-nearest-even
    return (unsigned short)(u >> 16);
}

static constexpr int N_NODES = 50000;
static constexpr int N_EDGES = 800000;

// LDS row strides (in bf16 elements): 16B aligned, bank-spread.
static constexpr int ASTR  = 40;
static constexpr int WSTR  = 40;
static constexpr int ASTR2 = 136;

// ---------------------------------------------------------------------------
// Weight prep: W[K x N] f32 row-major  ->  Wbf[n][Kpad] bf16, zero padded.
// ---------------------------------------------------------------------------
__global__ __launch_bounds__(256)
void prep_wbf_kernel(const float* __restrict__ W, int K, int N, int Kpad,
                     unsigned short* __restrict__ out)
{
    int idx = blockIdx.x * 256 + threadIdx.x;
    int total = N * Kpad;
    if (idx >= total) return;
    int n = idx / Kpad, k = idx - n * Kpad;
    out[idx] = (k < K) ? f2bf(W[(size_t)k * N + n]) : (unsigned short)0;
}

// ---------------------------------------------------------------------------
// Generic staged GEMM: out[M,N] = act(A[M,K] @ W + b), N in {64,128}.
// MODE 0: A = src.A (f32 row-major, lda, K arbitrary)          [scalar stage]
// MODE 1: A row = [hbf(128 bf16) | aggr(128 f32)]  (K=256)     [paired stage]
// MODE 2: A row = [coords_norm(3) | hbf(128)]      (K=131)     [scalar stage]
// MODE 3: A = src.Abf (bf16 row-major, 128 wide, K=128)        [paired stage]
// Weights come pre-transposed bf16: Wbf[n][Kpad].
// ---------------------------------------------------------------------------
struct GemmSrc {
    const float*          A;
    const unsigned short* Abf;
    const unsigned short* hbf;
    const float*          aggr;
    const float*          coords;
    const float*          cmm;   // [0..2]=min, [3..5]=max
    int lda;
    int K;
};

template <int MODE>
__device__ __forceinline__ unsigned short stage_u16(int row, int c, const GemmSrc& s) {
    if (MODE == 0) {
        return (c < s.K) ? f2bf(s.A[(size_t)row * s.lda + c]) : (unsigned short)0;
    } else { // MODE 2
        if (c < 3) {
            float mn = s.cmm[c], mx = s.cmm[3 + c];
            float rg = fmaxf(mx - mn, 1e-8f);
            return f2bf((s.coords[(size_t)row * 3 + c] - mn) / rg);
        } else if (c < 131) {
            return s.hbf[(size_t)row * 128 + (c - 3)];
        }
        return 0;
    }
}

template <int MODE>
__device__ __forceinline__ unsigned stage_u32(int row, int c, const GemmSrc& s) {
    if (MODE == 1) {
        if (c < 128) {
            return ((const unsigned*)(s.hbf + (size_t)row * 128))[c >> 1];
        } else {
            float a = s.aggr[(size_t)row * 128 + (c - 128)];
            float b = s.aggr[(size_t)row * 128 + (c - 127)];
            return (unsigned)f2bf(a) | ((unsigned)f2bf(b) << 16);
        }
    } else { // MODE 3
        return ((const unsigned*)(s.Abf + (size_t)row * 128))[c >> 1];
    }
}

template <int N, int MODE>
__global__ __launch_bounds__(256, 1)
void gemm_wmma_kernel(int M, GemmSrc src,
                      const unsigned short* __restrict__ Wbf, int Kpad,
                      const float* __restrict__ bias, int relu,
                      float* __restrict__ outf, unsigned short* __restrict__ outb)
{
    __shared__ unsigned short As[128 * ASTR];
    __shared__ unsigned short Ws[128 * WSTR];

    constexpr int  NT     = N >> 4;
    constexpr bool PAIRED = (MODE == 1) || (MODE == 3);
    const int tid  = threadIdx.x;
    const int lane = tid & 31;
    const int wave = tid >> 5;
    const int hi   = (lane >> 4) & 1;
    const int l16  = lane & 15;
    const int row0 = blockIdx.x * 128;
    const int nChunks = Kpad >> 5;

    v8f acc[NT];
    #pragma unroll
    for (int t = 0; t < NT; ++t)
        #pragma unroll
        for (int e = 0; e < 8; ++e) acc[t][e] = 0.f;

    for (int ch = 0; ch < nChunks; ++ch) {
        const int kc = ch << 5;
        // stage A chunk (128 rows x 32 k)
        if (PAIRED) {
            for (int i = tid; i < 128 * 16; i += 256) {
                int r = i >> 4, kp = i & 15;
                int gr = row0 + r, c = kc + kp * 2;
                unsigned val = (gr < M) ? stage_u32<MODE>(gr, c, src) : 0u;
                *(unsigned*)&As[r * ASTR + kp * 2] = val;
            }
        } else {
            for (int i = tid; i < 128 * 32; i += 256) {
                int r = i >> 5, kk = i & 31;
                int gr = row0 + r, c = kc + kk;
                As[r * ASTR + kk] = (gr < M) ? stage_u16<MODE>(gr, c, src) : (unsigned short)0;
            }
        }
        // stage W chunk: straight u32 copy from pre-transposed bf16 weights
        for (int i = tid; i < 16 * N; i += 256) {
            int n = i >> 4, kp = i & 15;
            unsigned val = *(const unsigned*)&Wbf[(size_t)n * Kpad + kc + kp * 2];
            *(unsigned*)&Ws[n * WSTR + kp * 2] = val;
        }
        __syncthreads();

        if (ch + 1 < nChunks) {   // prefetch next bf16 weight chunk (32*N*2 B)
            const char* pf = (const char*)Wbf + (size_t)(kc + 32) * 2;
            __builtin_prefetch(pf + (size_t)(tid >> 1) * Kpad * 2, 0, 1);
        }

        FragBF a;
        const u32x4* pa = (const u32x4*)&As[(wave * 16 + l16) * ASTR + hi * 8];
        a.q[0] = pa[0]; a.q[1] = pa[2];
        FragBF b[NT];
        #pragma unroll
        for (int t = 0; t < NT; ++t) {
            const u32x4* pb = (const u32x4*)&Ws[(t * 16 + l16) * WSTR + hi * 8];
            b[t].q[0] = pb[0]; b[t].q[1] = pb[2];
        }
        #pragma unroll
        for (int t = 0; t < NT; ++t) {
            acc[t] = __builtin_amdgcn_wmma_f32_16x16x32_bf16(
                false, a.v, false, b[t].v, (short)0, acc[t], false, false);
        }
        __syncthreads();
    }

    #pragma unroll
    for (int t = 0; t < NT; ++t) {
        int col = t * 16 + l16;
        float bv = bias[col];
        #pragma unroll
        for (int j = 0; j < 8; ++j) {
            int row = row0 + wave * 16 + j + hi * 8;
            if (row < M) {
                float v = acc[t][j] + bv;
                if (relu) v = fmaxf(v, 0.f);
                if (outf) outf[(size_t)row * N + col] = v;
                if (outb) outb[(size_t)row * N + col] = f2bf(v);
            }
        }
    }
}

// ---------------------------------------------------------------------------
// Fused 2-layer edge MLP. 128 edges per block, 8 waves.
// MODE 0 (ENC): in = edge_attr (E x 10 f32), out = bf16 e-buffer (E x 128)
// MODE 1 (MSG): in = [hbf[src] | hbf[dst] | e | rel] (E x 387),
//               out = scatter-add into aggr[dst] (global_atomic_add_f32)
// Weights pre-transposed bf16.
// ---------------------------------------------------------------------------
template <int MODE>
__global__ __launch_bounds__(256, 1)
void edge_mlp2_kernel(const unsigned short* __restrict__ hbf,
                      const unsigned short* __restrict__ ebuf,
                      const float* __restrict__ edge_attr,
                      const float* __restrict__ coords,
                      const int* __restrict__ src, const int* __restrict__ dst,
                      const unsigned short* __restrict__ W1bf, const float* __restrict__ b1,
                      const unsigned short* __restrict__ W2bf, const float* __restrict__ b2,
                      float* __restrict__ aggr,
                      unsigned short* __restrict__ eout,
                      int E)
{
    __shared__ unsigned short As[128 * ASTR];
    __shared__ unsigned short Ws[128 * WSTR];
    __shared__ unsigned short As2[128 * ASTR2];
    __shared__ int sidx[128];
    __shared__ int didx[128];

    const int tid  = threadIdx.x;
    const int lane = tid & 31;
    const int wave = tid >> 5;
    const int hi   = (lane >> 4) & 1;
    const int l16  = lane & 15;
    const int e0   = blockIdx.x * 128;

    if (tid < 128) { sidx[tid] = src[e0 + tid]; didx[tid] = dst[e0 + tid]; }
    __syncthreads();

    constexpr int Kpad1 = (MODE == 1) ? 416 : 32;
    constexpr int nCh   = Kpad1 >> 5;

    v8f acc[8];
    #pragma unroll
    for (int t = 0; t < 8; ++t)
        #pragma unroll
        for (int e = 0; e < 8; ++e) acc[t][e] = 0.f;

    // ---------------- layer 1: K1 -> 128, ReLU ----------------
    #pragma unroll 1
    for (int ch = 0; ch < nCh; ++ch) {
        const int kc = ch << 5;
        if (MODE == 1) {
            // paired staging: 2 bf16 per u32
            for (int i = tid; i < 128 * 16; i += 256) {
                int r = i >> 4, kp = i & 15, c = kc + kp * 2;
                unsigned val;
                if (c < 128)
                    val = ((const unsigned*)(hbf + (size_t)sidx[r] * 128))[c >> 1];
                else if (c < 256)
                    val = ((const unsigned*)(hbf + (size_t)didx[r] * 128))[(c - 128) >> 1];
                else if (c < 384)
                    val = ((const unsigned*)(ebuf + (size_t)(e0 + r) * 128))[(c - 256) >> 1];
                else if (c == 384) {
                    float r0 = coords[(size_t)didx[r] * 3 + 0] - coords[(size_t)sidx[r] * 3 + 0];
                    float r1 = coords[(size_t)didx[r] * 3 + 1] - coords[(size_t)sidx[r] * 3 + 1];
                    val = (unsigned)f2bf(r0) | ((unsigned)f2bf(r1) << 16);
                } else if (c == 386) {
                    float r2 = coords[(size_t)didx[r] * 3 + 2] - coords[(size_t)sidx[r] * 3 + 2];
                    val = (unsigned)f2bf(r2);
                } else val = 0u;
                *(unsigned*)&As[r * ASTR + kp * 2] = val;
            }
        } else {
            for (int i = tid; i < 128 * 32; i += 256) {
                int r = i >> 5, kk = i & 31, c = kc + kk;
                As[r * ASTR + kk] =
                    (c < 10) ? f2bf(edge_attr[(size_t)(e0 + r) * 10 + c]) : (unsigned short)0;
            }
        }
        for (int i = tid; i < 16 * 128; i += 256) {
            int n = i >> 4, kp = i & 15;
            unsigned val = *(const unsigned*)&W1bf[(size_t)n * Kpad1 + kc + kp * 2];
            *(unsigned*)&Ws[n * WSTR + kp * 2] = val;
        }
        __syncthreads();

        FragBF a;
        const u32x4* pa = (const u32x4*)&As[(wave * 16 + l16) * ASTR + hi * 8];
        a.q[0] = pa[0]; a.q[1] = pa[2];
        FragBF b[8];
        #pragma unroll
        for (int t = 0; t < 8; ++t) {
            const u32x4* pb = (const u32x4*)&Ws[(t * 16 + l16) * WSTR + hi * 8];
            b[t].q[0] = pb[0]; b[t].q[1] = pb[2];
        }
        #pragma unroll
        for (int t = 0; t < 8; ++t) {
            acc[t] = __builtin_amdgcn_wmma_f32_16x16x32_bf16(
                false, a.v, false, b[t].v, (short)0, acc[t], false, false);
        }
        __syncthreads();
    }

    // bias + ReLU -> bf16 intermediate in LDS (A-layout for layer 2)
    #pragma unroll
    for (int t = 0; t < 8; ++t) {
        int col = t * 16 + l16;
        float bv = b1[col];
        #pragma unroll
        for (int j = 0; j < 8; ++j) {
            int r = wave * 16 + j + hi * 8;
            float v = fmaxf(acc[t][j] + bv, 0.f);
            As2[r * ASTR2 + col] = f2bf(v);
        }
    }
    #pragma unroll
    for (int t = 0; t < 8; ++t)
        #pragma unroll
        for (int e = 0; e < 8; ++e) acc[t][e] = 0.f;
    __syncthreads();

    // ---------------- layer 2: 128 -> 128, linear ----------------
    #pragma unroll 1
    for (int ch = 0; ch < 4; ++ch) {
        const int kc = ch << 5;
        for (int i = tid; i < 16 * 128; i += 256) {
            int n = i >> 4, kp = i & 15;
            unsigned val = *(const unsigned*)&W2bf[(size_t)n * 128 + kc + kp * 2];
            *(unsigned*)&Ws[n * WSTR + kp * 2] = val;
        }
        __syncthreads();

        FragBF a;
        const u32x4* pa = (const u32x4*)&As2[(wave * 16 + l16) * ASTR2 + kc + hi * 8];
        a.q[0] = pa[0]; a.q[1] = pa[2];
        FragBF b[8];
        #pragma unroll
        for (int t = 0; t < 8; ++t) {
            const u32x4* pb = (const u32x4*)&Ws[(t * 16 + l16) * WSTR + hi * 8];
            b[t].q[0] = pb[0]; b[t].q[1] = pb[2];
        }
        #pragma unroll
        for (int t = 0; t < 8; ++t) {
            acc[t] = __builtin_amdgcn_wmma_f32_16x16x32_bf16(
                false, a.v, false, b[t].v, (short)0, acc[t], false, false);
        }
        __syncthreads();
    }

    // epilogue: scatter-add (MSG) or store bf16 (ENC)
    #pragma unroll
    for (int t = 0; t < 8; ++t) {
        int col = t * 16 + l16;
        float bv = b2[col];
        #pragma unroll
        for (int j = 0; j < 8; ++j) {
            int r = wave * 16 + j + hi * 8;
            float v = acc[t][j] + bv;
            if (MODE == 1) {
                unsafeAtomicAdd(&aggr[(size_t)didx[r] * 128 + col], v);  // global_atomic_add_f32
            } else {
                eout[(size_t)(e0 + r) * 128 + col] = f2bf(v);
            }
        }
    }
}

// ---------------------------------------------------------------------------
// coords min/max reduction (single block)
// ---------------------------------------------------------------------------
__global__ __launch_bounds__(256)
void minmax_kernel(const float* __restrict__ coords, int n, float* __restrict__ cmm)
{
    __shared__ float smn[256 * 3];
    __shared__ float smx[256 * 3];
    int tid = threadIdx.x;
    float mn[3] = { 3.4e38f, 3.4e38f, 3.4e38f };
    float mx[3] = { -3.4e38f, -3.4e38f, -3.4e38f };
    for (int i = tid; i < n; i += 256) {
        #pragma unroll
        for (int c = 0; c < 3; ++c) {
            float v = coords[(size_t)i * 3 + c];
            mn[c] = fminf(mn[c], v);
            mx[c] = fmaxf(mx[c], v);
        }
    }
    #pragma unroll
    for (int c = 0; c < 3; ++c) { smn[tid * 3 + c] = mn[c]; smx[tid * 3 + c] = mx[c]; }
    __syncthreads();
    for (int s = 128; s > 0; s >>= 1) {
        if (tid < s) {
            #pragma unroll
            for (int c = 0; c < 3; ++c) {
                smn[tid * 3 + c] = fminf(smn[tid * 3 + c], smn[(tid + s) * 3 + c]);
                smx[tid * 3 + c] = fmaxf(smx[tid * 3 + c], smx[(tid + s) * 3 + c]);
            }
        }
        __syncthreads();
    }
    if (tid < 3) { cmm[tid] = smn[tid]; cmm[3 + tid] = smx[tid]; }
}

// ---------------------------------------------------------------------------
// LayerNorm(h + u) -> h (f32) and hbf (bf16 mirror). One wave per node.
// ---------------------------------------------------------------------------
__global__ __launch_bounds__(256)
void ln_kernel(const float* __restrict__ h, const float* __restrict__ u,
               const float* __restrict__ gamma, const float* __restrict__ beta,
               float* __restrict__ out, unsigned short* __restrict__ outb, int M)
{
    int lane = threadIdx.x & 31;
    int wave = threadIdx.x >> 5;
    int node = blockIdx.x * 8 + wave;
    if (node >= M) return;

    const float4 a = ((const float4*)(h + (size_t)node * 128))[lane];
    const float4 b = ((const float4*)(u + (size_t)node * 128))[lane];
    float v0 = a.x + b.x, v1 = a.y + b.y, v2 = a.z + b.z, v3 = a.w + b.w;

    float s = v0 + v1 + v2 + v3;
    for (int m = 16; m > 0; m >>= 1) s += __shfl_xor(s, m, 32);
    float mu = s * (1.f / 128.f);

    float d0 = v0 - mu, d1 = v1 - mu, d2 = v2 - mu, d3 = v3 - mu;
    float q = d0 * d0 + d1 * d1 + d2 * d2 + d3 * d3;
    for (int m = 16; m > 0; m >>= 1) q += __shfl_xor(q, m, 32);
    float rstd = rsqrtf(q * (1.f / 128.f) + 1e-5f);

    const float4 g  = ((const float4*)gamma)[lane];
    const float4 be = ((const float4*)beta)[lane];
    float4 o;
    o.x = g.x * d0 * rstd + be.x;
    o.y = g.y * d1 * rstd + be.y;
    o.z = g.z * d2 * rstd + be.z;
    o.w = g.w * d3 * rstd + be.w;
    ((float4*)(out + (size_t)node * 128))[lane] = o;

    unsigned p0 = (unsigned)f2bf(o.x) | ((unsigned)f2bf(o.y) << 16);
    unsigned p1 = (unsigned)f2bf(o.z) | ((unsigned)f2bf(o.w) << 16);
    unsigned* hb = (unsigned*)(outb + (size_t)node * 128);
    hb[lane * 2 + 0] = p0;
    hb[lane * 2 + 1] = p1;
}

// ---------------------------------------------------------------------------
// Decoder tail: y = x64 @ W3(64x3) + b3, then BC masks -> d_out (M x 3)
// ---------------------------------------------------------------------------
__global__ __launch_bounds__(256)
void dec_final_kernel(const float* __restrict__ a64,
                      const float* __restrict__ W3, const float* __restrict__ b3,
                      const float* __restrict__ bcd, const float* __restrict__ bcr,
                      float* __restrict__ out, int M)
{
    int n = blockIdx.x * blockDim.x + threadIdx.x;
    if (n >= M) return;
    float p0 = b3[0], p1 = b3[1], p2 = b3[2];
    const float* xr = a64 + (size_t)n * 64;
    #pragma unroll 8
    for (int k = 0; k < 64; ++k) {
        float v = xr[k];
        p0 += v * W3[k * 3 + 0];
        p1 += v * W3[k * 3 + 1];
        p2 += v * W3[k * 3 + 2];
    }
    float md = 1.f - bcd[n];
    float mr = 1.f - bcr[n];
    out[(size_t)n * 3 + 0] = p0 * md;
    out[(size_t)n * 3 + 1] = p1 * md;
    out[(size_t)n * 3 + 2] = p2 * mr;
}

// ---------------------------------------------------------------------------
// Host orchestration
// ---------------------------------------------------------------------------
extern "C" void kernel_launch(void* const* d_in, const int* in_sizes, int n_in,
                              void* d_out, int out_size, void* d_ws, size_t ws_size,
                              hipStream_t stream)
{
    const float* x         = (const float*)d_in[0];
    const float* edge_attr = (const float*)d_in[1];
    const float* coords    = (const float*)d_in[2];
    const float* bcd       = (const float*)d_in[3];
    const float* bcr       = (const float*)d_in[4];
    const int*   ei        = (const int*)d_in[5];
    const int*   srcI      = ei;
    const int*   dstI      = ei + N_EDGES;

    // Param leaf mapping: handle both insertion-order and sorted-key pytree
    // flattening; distinguish via in_sizes[6].
    const float *neW[2], *neB[2], *eeW[2], *eeB[2], *dW[4], *dB[4];
    const float *mW0[6], *mB0[6], *mW1[6], *mB1[6];
    const float *uW0[6], *uB0[6], *uW1[6], *uB1[6], *gam[6], *bet[6];

    bool insertion = (in_sizes[6] == 9 * 128);
    if (insertion) {
        neW[0] = (const float*)d_in[6];  neB[0] = (const float*)d_in[7];
        neW[1] = (const float*)d_in[8];  neB[1] = (const float*)d_in[9];
        eeW[0] = (const float*)d_in[10]; eeB[0] = (const float*)d_in[11];
        eeW[1] = (const float*)d_in[12]; eeB[1] = (const float*)d_in[13];
        for (int i = 0; i < 4; ++i) {
            dW[i] = (const float*)d_in[14 + 2 * i];
            dB[i] = (const float*)d_in[15 + 2 * i];
        }
        for (int l = 0; l < 6; ++l) {
            int b = 22 + l * 10;
            mW0[l] = (const float*)d_in[b + 0]; mB0[l] = (const float*)d_in[b + 1];
            mW1[l] = (const float*)d_in[b + 2]; mB1[l] = (const float*)d_in[b + 3];
            uW0[l] = (const float*)d_in[b + 4]; uB0[l] = (const float*)d_in[b + 5];
            uW1[l] = (const float*)d_in[b + 6]; uB1[l] = (const float*)d_in[b + 7];
            gam[l] = (const float*)d_in[b + 8]; bet[l] = (const float*)d_in[b + 9];
        }
    } else { // sorted keys: decoder, edge_enc, layers(beta,gamma,msg,upd), node_enc
        for (int i = 0; i < 4; ++i) {
            dW[i] = (const float*)d_in[6 + 2 * i];
            dB[i] = (const float*)d_in[7 + 2 * i];
        }
        eeW[0] = (const float*)d_in[14]; eeB[0] = (const float*)d_in[15];
        eeW[1] = (const float*)d_in[16]; eeB[1] = (const float*)d_in[17];
        for (int l = 0; l < 6; ++l) {
            int b = 18 + l * 10;
            bet[l] = (const float*)d_in[b + 0]; gam[l] = (const float*)d_in[b + 1];
            mW0[l] = (const float*)d_in[b + 2]; mB0[l] = (const float*)d_in[b + 3];
            mW1[l] = (const float*)d_in[b + 4]; mB1[l] = (const float*)d_in[b + 5];
            uW0[l] = (const float*)d_in[b + 6]; uB0[l] = (const float*)d_in[b + 7];
            uW1[l] = (const float*)d_in[b + 8]; uB1[l] = (const float*)d_in[b + 9];
        }
        neW[0] = (const float*)d_in[78]; neB[0] = (const float*)d_in[79];
        neW[1] = (const float*)d_in[80]; neB[1] = (const float*)d_in[81];
    }

    // workspace carve-up
    char* ws = (char*)d_ws;
    size_t off = 0;
    auto carve = [&](size_t bytes) -> char* {
        char* p = ws + off;
        off = (off + bytes + 255) & ~(size_t)255;
        return p;
    };
    float*          cmm   = (float*)carve(64 * sizeof(float));
    unsigned short* ebuf  = (unsigned short*)carve((size_t)N_EDGES * 128 * 2);
    float*          h     = (float*)carve((size_t)N_NODES * 128 * 4);
    unsigned short* hbf   = (unsigned short*)carve((size_t)N_NODES * 128 * 2);
    float*          aggr  = (float*)carve((size_t)N_NODES * 128 * 4);
    float*          ubuf  = (float*)carve((size_t)N_NODES * 128 * 4);
    unsigned short* t0b   = (unsigned short*)carve((size_t)N_NODES * 128 * 2);
    unsigned short* t1b   = (unsigned short*)carve((size_t)N_NODES * 128 * 2);
    float*          dec64 = (float*)carve((size_t)N_NODES * 64 * 4);

    // pre-transposed bf16 weights
    unsigned short* ne0b = (unsigned short*)carve(128 * 32 * 2);
    unsigned short* ne1b = (unsigned short*)carve(128 * 128 * 2);
    unsigned short* ee0b = (unsigned short*)carve(128 * 32 * 2);
    unsigned short* ee1b = (unsigned short*)carve(128 * 128 * 2);
    unsigned short *m1b[6], *m2b[6], *u0b[6], *u1b[6];
    for (int l = 0; l < 6; ++l) {
        m1b[l] = (unsigned short*)carve((size_t)128 * 416 * 2);
        m2b[l] = (unsigned short*)carve((size_t)128 * 128 * 2);
        u0b[l] = (unsigned short*)carve((size_t)128 * 256 * 2);
        u1b[l] = (unsigned short*)carve((size_t)128 * 128 * 2);
    }
    unsigned short* d0b = (unsigned short*)carve((size_t)128 * 160 * 2);
    unsigned short* d1b = (unsigned short*)carve((size_t)128 * 128 * 2);
    unsigned short* d2b = (unsigned short*)carve((size_t)64 * 128 * 2);

    dim3 blk(256);
    auto prep = [&](const float* W, int K, int N, int Kpad, unsigned short* dst) {
        int total = N * Kpad;
        prep_wbf_kernel<<<(total + 255) / 256, blk, 0, stream>>>(W, K, N, Kpad, dst);
    };
    prep(neW[0], 9, 128, 32, ne0b);
    prep(neW[1], 128, 128, 128, ne1b);
    prep(eeW[0], 10, 128, 32, ee0b);
    prep(eeW[1], 128, 128, 128, ee1b);
    for (int l = 0; l < 6; ++l) {
        prep(mW0[l], 387, 128, 416, m1b[l]);
        prep(mW1[l], 128, 128, 128, m2b[l]);
        prep(uW0[l], 256, 128, 256, u0b[l]);
        prep(uW1[l], 128, 128, 128, u1b[l]);
    }
    prep(dW[0], 131, 128, 160, d0b);
    prep(dW[1], 128, 128, 128, d1b);
    prep(dW[2], 128, 64, 128, d2b);

    const int gn = (N_NODES + 127) / 128;   // 391
    const int ge = N_EDGES / 128;           // 6250 (exact)

    minmax_kernel<<<1, blk, 0, stream>>>(coords, N_NODES, cmm);

    GemmSrc s{};
    // node encoder: 9 -> 128 (ReLU, bf16 only) -> 128 (f32 + bf16 mirror)
    s = GemmSrc{}; s.A = x; s.lda = 9; s.K = 9;
    gemm_wmma_kernel<128, 0><<<gn, blk, 0, stream>>>(N_NODES, s, ne0b, 32, neB[0], 1, nullptr, t0b);
    s = GemmSrc{}; s.Abf = t0b;
    gemm_wmma_kernel<128, 3><<<gn, blk, 0, stream>>>(N_NODES, s, ne1b, 128, neB[1], 0, h, hbf);

    // edge encoder: 10 -> 128 (ReLU) -> 128, output bf16
    edge_mlp2_kernel<0><<<ge, blk, 0, stream>>>(nullptr, nullptr, edge_attr, nullptr,
                                                srcI, dstI, ee0b, eeB[0], ee1b, eeB[1],
                                                nullptr, ebuf, N_EDGES);

    // 12 message-passing passes (each of the 6 layers applied twice)
    for (int it = 0; it < 12; ++it) {
        int l = it >> 1;
        hipMemsetAsync(aggr, 0, (size_t)N_NODES * 128 * 4, stream);
        edge_mlp2_kernel<1><<<ge, blk, 0, stream>>>(hbf, ebuf, nullptr, coords,
                                                    srcI, dstI, m1b[l], mB0[l], m2b[l], mB1[l],
                                                    aggr, nullptr, N_EDGES);
        s = GemmSrc{}; s.hbf = hbf; s.aggr = aggr;
        gemm_wmma_kernel<128, 1><<<gn, blk, 0, stream>>>(N_NODES, s, u0b[l], 256, uB0[l], 1, nullptr, t0b);
        s = GemmSrc{}; s.Abf = t0b;
        gemm_wmma_kernel<128, 3><<<gn, blk, 0, stream>>>(N_NODES, s, u1b[l], 128, uB1[l], 0, ubuf, nullptr);
        ln_kernel<<<(N_NODES + 7) / 8, blk, 0, stream>>>(h, ubuf, gam[l], bet[l], h, hbf, N_NODES);
    }

    // decoder: [coords_norm|h](131) -> 128 -> 128 -> 64 -> 3, masks
    s = GemmSrc{}; s.hbf = hbf; s.coords = coords; s.cmm = cmm; s.K = 131;
    gemm_wmma_kernel<128, 2><<<gn, blk, 0, stream>>>(N_NODES, s, d0b, 160, dB[0], 1, nullptr, t0b);
    s = GemmSrc{}; s.Abf = t0b;
    gemm_wmma_kernel<128, 3><<<gn, blk, 0, stream>>>(N_NODES, s, d1b, 128, dB[1], 1, nullptr, t1b);
    s = GemmSrc{}; s.Abf = t1b;
    gemm_wmma_kernel<64, 3><<<gn, blk, 0, stream>>>(N_NODES, s, d2b, 128, dB[2], 1, dec64, nullptr);
    dec_final_kernel<<<(N_NODES + 255) / 256, blk, 0, stream>>>(dec64, dW[3], dB[3],
                                                                bcd, bcr, (float*)d_out, N_NODES);
}
